// SpatialAttention_89361089561496
// MI455X (gfx1250) — matrix operand order repaired and checked by reading, not compiled, not run on hardware
//
#include <hip/hip_runtime.h>
#include <hip/hip_bf16.h>

// ---------------------------------------------------------------------------
// SpatialAttention for MI455X (gfx1250, wave32, WMMA 16x16x32 f16 -> f32 acc)
//   x:      (1, 256, 16,16,16) fp32  -> n=4096 tokens, c=256 channels
//   w_qkv:  (768, 256), w_out: (256, 256), b_out: (256)
//   8 heads x dh=32, flash-attention (online softmax), everything L2-resident
// ---------------------------------------------------------------------------

typedef __attribute__((ext_vector_type(16))) _Float16 v16h;
typedef __attribute__((ext_vector_type(8)))  _Float16 v8h;
typedef __attribute__((ext_vector_type(8)))  float    v8f;

union V16 { v16h v; v8h h[2]; };

#define N_TOK   4096
#define CDIM    256
#define HEADS   8
#define DH      32
#define HIDDEN  256
#define SCALE_F 0.17677669529663687f   // 32^-0.5

__device__ __forceinline__ v8f wmma_f16(v16h a, v16h b, v8f c) {
  // (neg_a, A, neg_b, B, c_mod, C, reuse_a, reuse_b)
  return __builtin_amdgcn_wmma_f32_16x16x32_f16(false, a, false, b, (short)0, c,
                                                false, false);
}

// --------------------------------------------------------------------------
// Kernel 0: fp32 -> f16 conversions.  x [256][4096] -> xT [4096][256] f16.
// --------------------------------------------------------------------------
__global__ __launch_bounds__(256) void convert_kernel(
    const float* __restrict__ x, const float* __restrict__ wqkv,
    const float* __restrict__ wout,
    _Float16* __restrict__ xT, _Float16* __restrict__ wqkv_h,
    _Float16* __restrict__ wout_h) {
  int idx = blockIdx.x * blockDim.x + threadIdx.x;
  if (idx < CDIM * N_TOK) {
    int c = idx >> 12;          // / 4096
    int i = idx & (N_TOK - 1);
    xT[i * CDIM + c] = (_Float16)x[idx];
  }
  if (idx < 3 * HIDDEN * CDIM) wqkv_h[idx] = (_Float16)wqkv[idx];
  if (idx < CDIM * HIDDEN)     wout_h[idx] = (_Float16)wout[idx];
}

// --------------------------------------------------------------------------
// Kernel 1: QKV GEMM  (768x4096 = wqkv[768x256] * x[256x4096]) via WMMA.
// One wave per 16x16 output tile, K-loop of 8 WMMAs.
// Writes Qt/Kt as [h][n][32] f16 (Q pre-scaled), V as [h][32][n] f16.
// --------------------------------------------------------------------------
__global__ __launch_bounds__(128) void qkv_gemm_kernel(
    const _Float16* __restrict__ wq,   // [768][256]
    const _Float16* __restrict__ xT,   // [4096][256]
    _Float16* __restrict__ Qt, _Float16* __restrict__ Kt,
    _Float16* __restrict__ Vc) {
  const int lane = threadIdx.x & 31;
  const int lo = lane & 15, g = lane >> 4;
  const int wid = blockIdx.x * 4 + (threadIdx.x >> 5);   // 0..12287
  const int otile = wid >> 8;                            // 0..47
  const int itile = wid & 255;                           // 0..255
  const int o0 = otile * 16, i0 = itile * 16;

  const _Float16* arow = wq + (size_t)(o0 + lo) * CDIM;
  const _Float16* brow = xT + (size_t)(i0 + lo) * CDIM;

  v8f acc = {};
#pragma unroll
  for (int k0 = 0; k0 < CDIM; k0 += 32) {
    V16 a, b;
    a.h[0] = *(const v8h*)(arow + k0 + 8 * g);
    a.h[1] = *(const v8h*)(arow + k0 + 16 + 8 * g);
    b.h[0] = *(const v8h*)(brow + k0 + 8 * g);
    b.h[1] = *(const v8h*)(brow + k0 + 16 + 8 * g);
    acc = wmma_f16(a.v, b.v, acc);
  }

#pragma unroll
  for (int r = 0; r < 8; ++r) {
    int o = o0 + r + 8 * g;            // output channel 0..767
    int i = i0 + lo;                   // token
    float val = acc[r];
    if (o < HIDDEN) {                  // Q (pre-scaled)
      int h = o >> 5, c = o & 31;
      Qt[((size_t)h * N_TOK + i) * DH + c] = (_Float16)(val * SCALE_F);
    } else if (o < 2 * HIDDEN) {       // K
      int oo = o - HIDDEN;
      int h = oo >> 5, c = oo & 31;
      Kt[((size_t)h * N_TOK + i) * DH + c] = (_Float16)val;
    } else {                           // V, kept channel-major
      int oo = o - 2 * HIDDEN;
      Vc[(size_t)oo * N_TOK + i] = (_Float16)val;
    }
  }
}

// --------------------------------------------------------------------------
// Kernel 2: flash attention.  One wave handles (head, 16-query block),
// streaming 32 keys/iter: 2 WMMAs for S = Qt^T*K, online softmax,
// D->A transpose of P through a 1KB per-wave LDS tile, 2 WMMAs for O += P*V.
// Output: attnT [n=4096][hc=256] f16.
// --------------------------------------------------------------------------
__global__ __launch_bounds__(128) void attn_kernel(
    const _Float16* __restrict__ Qt, const _Float16* __restrict__ Kt,
    const _Float16* __restrict__ Vc, _Float16* __restrict__ attnT) {
  __shared__ _Float16 lds[4][16 * 32];
  const int lane = threadIdx.x & 31;
  const int lo = lane & 15, g = lane >> 4;
  const int widx = threadIdx.x >> 5;
  const int wid = blockIdx.x * 4 + widx;     // 0..2047
  const int h = wid >> 8;                    // head 0..7
  const int i0 = (wid & 255) * 16;           // query block base
  _Float16* P = lds[widx];

  // Q tile in A layout: row i = i0+lo, K = channel
  const _Float16* qrow = Qt + ((size_t)h * N_TOK + i0 + lo) * DH;
  V16 qa;
  qa.h[0] = *(const v8h*)(qrow + 8 * g);
  qa.h[1] = *(const v8h*)(qrow + 16 + 8 * g);

  const _Float16* krowbase = Kt + (size_t)h * N_TOK * DH;
  const _Float16* v0row = Vc + ((size_t)h * DH + lo) * N_TOK;        // c-tile 0
  const _Float16* v1row = v0row + (size_t)16 * N_TOK;                // c-tile 1

  float mrow[8], lrow[8];
  v8f oa0 = {}, oa1 = {};
#pragma unroll
  for (int r = 0; r < 8; ++r) { mrow[r] = -1e30f; lrow[r] = 0.f; }

  for (int j0 = 0; j0 < N_TOK; j0 += 32) {
    // ---- S tiles: D = Qt^T (A) x K (B) ----
    const _Float16* kr0 = krowbase + (size_t)(j0 + lo) * DH;
    const _Float16* kr1 = krowbase + (size_t)(j0 + 16 + lo) * DH;
    V16 kb0, kb1;
    kb0.h[0] = *(const v8h*)(kr0 + 8 * g);
    kb0.h[1] = *(const v8h*)(kr0 + 16 + 8 * g);
    kb1.h[0] = *(const v8h*)(kr1 + 8 * g);
    kb1.h[1] = *(const v8h*)(kr1 + 16 + 8 * g);
    v8f zero = {};
    v8f s0 = wmma_f16(qa.v, kb0.v, zero);
    v8f s1 = wmma_f16(qa.v, kb1.v, zero);

    // ---- online softmax: row max across 16 lanes of each half-wave ----
    float nm[8];
#pragma unroll
    for (int r = 0; r < 8; ++r) nm[r] = fmaxf(s0[r], s1[r]);
#pragma unroll
    for (int off = 1; off <= 8; off <<= 1)
#pragma unroll
      for (int r = 0; r < 8; ++r) nm[r] = fmaxf(nm[r], __shfl_xor(nm[r], off, 32));

    float corr[8];
#pragma unroll
    for (int r = 0; r < 8; ++r) {
      nm[r] = fmaxf(nm[r], mrow[r]);
      corr[r] = __expf(mrow[r] - nm[r]);
      mrow[r] = nm[r];
    }
    float rs[8];
#pragma unroll
    for (int r = 0; r < 8; ++r) {
      float p0 = __expf(s0[r] - mrow[r]);
      float p1 = __expf(s1[r] - mrow[r]);
      s0[r] = p0; s1[r] = p1;
      rs[r] = p0 + p1;
    }
#pragma unroll
    for (int off = 1; off <= 8; off <<= 1)
#pragma unroll
      for (int r = 0; r < 8; ++r) rs[r] += __shfl_xor(rs[r], off, 32);
#pragma unroll
    for (int r = 0; r < 8; ++r) {
      lrow[r] = lrow[r] * corr[r] + rs[r];
      oa0[r] *= corr[r];
      oa1[r] *= corr[r];
    }

    // ---- P: D layout -> A layout via per-wave LDS tile (f32->f16) ----
#pragma unroll
    for (int r = 0; r < 8; ++r) {
      P[(r + 8 * g) * 32 + lo]      = (_Float16)s0[r];
      P[(r + 8 * g) * 32 + 16 + lo] = (_Float16)s1[r];
    }
    const _Float16* prow = P + lo * 32;   // A row = query lo (same-wave DS: in order)
    V16 pa;
    pa.h[0] = *(const v8h*)(prow + 8 * g);
    pa.h[1] = *(const v8h*)(prow + 16 + 8 * g);

    // ---- O += P (A) x V (B), two c-tiles ----
    V16 vb0, vb1;
    vb0.h[0] = *(const v8h*)(v0row + j0 + 8 * g);
    vb0.h[1] = *(const v8h*)(v0row + j0 + 16 + 8 * g);
    vb1.h[0] = *(const v8h*)(v1row + j0 + 8 * g);
    vb1.h[1] = *(const v8h*)(v1row + j0 + 16 + 8 * g);
    oa0 = wmma_f16(pa.v, vb0.v, oa0);
    oa1 = wmma_f16(pa.v, vb1.v, oa1);
  }

  // ---- finalize: O/l, write attnT [i][hc] ----
#pragma unroll
  for (int r = 0; r < 8; ++r) {
    float inv = 1.0f / lrow[r];
    size_t i = (size_t)(i0 + r + 8 * g);
    attnT[i * HIDDEN + h * DH + lo]      = (_Float16)(oa0[r] * inv);
    attnT[i * HIDDEN + h * DH + 16 + lo] = (_Float16)(oa1[r] * inv);
  }
}

// --------------------------------------------------------------------------
// Kernel 3: output projection y = w_out[256x256] * attn[256x4096] + b, fp32.
// --------------------------------------------------------------------------
__global__ __launch_bounds__(128) void out_gemm_kernel(
    const _Float16* __restrict__ wo,   // [256][256] f16
    const _Float16* __restrict__ aT,   // [4096][256] f16
    const float* __restrict__ bias, float* __restrict__ y) {
  const int lane = threadIdx.x & 31;
  const int lo = lane & 15, g = lane >> 4;
  const int wid = blockIdx.x * 4 + (threadIdx.x >> 5);   // 0..4095
  const int otile = wid >> 8;                            // 0..15
  const int itile = wid & 255;
  const int o0 = otile * 16, i0 = itile * 16;

  const _Float16* arow = wo + (size_t)(o0 + lo) * HIDDEN;
  const _Float16* brow = aT + (size_t)(i0 + lo) * HIDDEN;

  v8f acc = {};
#pragma unroll
  for (int k0 = 0; k0 < HIDDEN; k0 += 32) {
    V16 a, b;
    a.h[0] = *(const v8h*)(arow + k0 + 8 * g);
    a.h[1] = *(const v8h*)(arow + k0 + 16 + 8 * g);
    b.h[0] = *(const v8h*)(brow + k0 + 8 * g);
    b.h[1] = *(const v8h*)(brow + k0 + 16 + 8 * g);
    acc = wmma_f16(a.v, b.v, acc);
  }

#pragma unroll
  for (int r = 0; r < 8; ++r) {
    int o = o0 + r + 8 * g;
    y[(size_t)o * N_TOK + i0 + lo] = acc[r] + bias[o];
  }
}

// --------------------------------------------------------------------------
// Host-side launch
// --------------------------------------------------------------------------
extern "C" void kernel_launch(void* const* d_in, const int* in_sizes, int n_in,
                              void* d_out, int out_size, void* d_ws, size_t ws_size,
                              hipStream_t stream) {
  (void)in_sizes; (void)n_in; (void)out_size; (void)ws_size;
  const float* x     = (const float*)d_in[0];   // 256*4096
  const float* wqkv  = (const float*)d_in[1];   // 768*256
  const float* wout  = (const float*)d_in[2];   // 256*256
  const float* bout  = (const float*)d_in[3];   // 256
  float* y = (float*)d_out;                     // 256*4096

  char* ws = (char*)d_ws;
  size_t off = 0;
  _Float16* xT     = (_Float16*)(ws + off); off += (size_t)N_TOK * CDIM * 2;       // 2 MB
  _Float16* wqkv_h = (_Float16*)(ws + off); off += (size_t)3 * HIDDEN * CDIM * 2;  // 384 KB
  _Float16* wout_h = (_Float16*)(ws + off); off += (size_t)CDIM * HIDDEN * 2;      // 128 KB
  _Float16* Qt     = (_Float16*)(ws + off); off += (size_t)HEADS * N_TOK * DH * 2; // 2 MB
  _Float16* Kt     = (_Float16*)(ws + off); off += (size_t)HEADS * N_TOK * DH * 2; // 2 MB
  _Float16* Vc     = (_Float16*)(ws + off); off += (size_t)HEADS * DH * N_TOK * 2; // 2 MB
  _Float16* attnT  = (_Float16*)(ws + off); off += (size_t)N_TOK * HIDDEN * 2;     // 2 MB

  // 0) fp32 -> f16 (+ transpose of x)
  convert_kernel<<<(CDIM * N_TOK + 255) / 256, 256, 0, stream>>>(
      x, wqkv, wout, xT, wqkv_h, wout_h);

  // 1) QKV projection: 48 x 256 tiles, 4 waves/block
  qkv_gemm_kernel<<<(48 * 256) / 4, 128, 0, stream>>>(wqkv_h, xT, Qt, Kt, Vc);

  // 2) flash attention: 8 heads x 256 query blocks, 4 waves/block
  attn_kernel<<<(HEADS * 256) / 4, 128, 0, stream>>>(Qt, Kt, Vc, attnT);

  // 3) output projection: 16 x 256 tiles, 4 waves/block
  out_gemm_kernel<<<(16 * 256) / 4, 128, 0, stream>>>(wout_h, attnT, bout, y);
}
